// FusionAttention_42236708389332
// MI455X (gfx1250) — matrix operand order repaired and checked by reading, not compiled, not run on hardware
//
#include <hip/hip_runtime.h>

typedef __attribute__((ext_vector_type(16))) _Float16 v16h;
typedef __attribute__((ext_vector_type(8)))  _Float16 v8h;
typedef __attribute__((ext_vector_type(8)))  float    v8f;

#define DEV static __device__ __forceinline__

constexpr int Bb = 2, Tt = 8, Ll = 1024, Cc = 256, Hh = 8, Dd = 32;
constexpr int BT   = Bb * Tt;   // 16
constexpr int Mtot = BT * Ll;   // 16384 rows for the projection GEMMs

// ---------------------------------------------------------------- WMMA helpers

DEV v8f wmma16(v16h a, v16h b, v8f c) {
  // D = A(16x32 f16) * B(32x16 f16) + C(16x16 f32)
  return __builtin_amdgcn_wmma_f32_16x16x32_f16(false, a, false, b, (short)0, c,
                                                false, false);
}

// A-fragment (16-bit A, 16x32): lane L holds row M=L%16;
// hl=L/16: VGPR0-3 <- K = k0+hl*16 .. +7, VGPR4-7 <- K = k0+hl*16+8 .. +7.
DEV v16h load_a_f32(const float* row, int k0, int hl) {
  const float* p0 = row + k0 + hl * 16;
  const float* p1 = row + k0 + hl * 16 + 8;
  v16h a;
#pragma unroll
  for (int j = 0; j < 8; ++j) { a[j] = (_Float16)p0[j]; a[8 + j] = (_Float16)p1[j]; }
  return a;
}

DEV v16h load_a_f16(const _Float16* row, int k0, int hl) {
  const v8h lo = *(const v8h*)(row + k0 + hl * 16);
  const v8h hi = *(const v8h*)(row + k0 + hl * 16 + 8);
  v16h a;
#pragma unroll
  for (int j = 0; j < 8; ++j) { a[j] = lo[j]; a[8 + j] = hi[j]; }
  return a;
}

// -------------------------------------------- kernel 0: weights fp32 -> f16
__global__ __launch_bounds__(256) void cvt_w_kernel(
    const float* __restrict__ w0, const float* __restrict__ w1,
    const float* __restrict__ w2, const float* __restrict__ w3,
    _Float16* __restrict__ o0, _Float16* __restrict__ o1,
    _Float16* __restrict__ o2, _Float16* __restrict__ o3) {
  const int i = blockIdx.x * 256 + threadIdx.x;   // grid covers Cc*Cc exactly
  o0[i] = (_Float16)w0[i];
  o1[i] = (_Float16)w1[i];
  o2[i] = (_Float16)w2[i];
  o3[i] = (_Float16)w3[i];
}

// ------------------------------------------------- kernel 1: proj + L2-norm
// y = x @ W^T + b (M=16384, N=K=256), then per (row, head) L2-normalize d=32.
// Block = 64 rows x 64 cols (2 whole heads); 8 waves = 4 M-tiles x 2 N-halves.
// mode 0: q -> [bt,h,l,d] f16, pre-scaled by 1/sqrt(d)
// mode 1: k -> [bt,h,l,d] f16
// mode 2: v -> [bt,h,d,l] f16 (transposed for the AV WMMA B-fragment)
__global__ __launch_bounds__(256) void proj_norm_kernel(
    const float* __restrict__ x, const _Float16* __restrict__ Wh,
    const float* __restrict__ bias, _Float16* __restrict__ out, int mode) {
  __shared__ float tile[64][64];

  const int wave = threadIdx.x >> 5;
  const int lane = threadIdx.x & 31;
  const int hl   = lane >> 4;
  const int l16  = lane & 15;
  const int mt   = wave >> 1;      // 0..3 : 16-row tile in block
  const int nh   = wave & 1;       // 0..1 : 32-col half in block
  const int mblk = blockIdx.x;     // 64 rows
  const int nblk = blockIdx.y;     // 64 cols

  const float* xrow = x + (size_t)(mblk * 64 + mt * 16 + l16) * Cc;
  const int n0 = nblk * 64 + nh * 32 + l16;

  v8f acc0 = {}, acc1 = {};
#pragma unroll
  for (int k0 = 0; k0 < Cc; k0 += 32) {
    const v16h a  = load_a_f32(xrow, k0, hl);
    const v16h b0 = *(const v16h*)(Wh + (size_t)n0 * Cc + k0 + hl * 16);
    const v16h b1 = *(const v16h*)(Wh + (size_t)(n0 + 16) * Cc + k0 + hl * 16);
    acc0 = wmma16(a, b0, acc0);
    acc1 = wmma16(a, b1, acc1);
  }

  // Stage C tiles (+bias) to LDS: C layout is N=l16, M=v+8*hl.
#pragma unroll
  for (int nt = 0; nt < 2; ++nt) {
    const int lc = nh * 32 + nt * 16 + l16;
    const float bv = bias[nblk * 64 + lc];
    const v8f acc = nt ? acc1 : acc0;
#pragma unroll
    for (int v = 0; v < 8; ++v) tile[mt * 16 + v + 8 * hl][lc] = acc[v] + bv;
  }
  __syncthreads();

  if (threadIdx.x < 128) {                 // 64 rows x 2 heads per block
    const int r  = threadIdx.x >> 1;       // 0..63
    const int hg = threadIdx.x & 1;        // head within the 64-col block
    float vals[Dd];
    float ss = 0.f;
#pragma unroll
    for (int j = 0; j < Dd; ++j) { vals[j] = tile[r][hg * Dd + j]; ss += vals[j] * vals[j]; }
    const float nrm = fmaxf(sqrtf(ss), 1e-12f);
    const float sc  = (mode == 0 ? 0.17677669529663687f : 1.0f) / nrm; // fold 1/sqrt(d)
    const int hh = nblk * 2 + hg;
    const int Mg = mblk * 64 + r;
    const int bt = Mg >> 10;               // / Ll
    const int lr = Mg & (Ll - 1);
    if (mode != 2) {
      _Float16* dst = out + ((size_t)(bt * Hh + hh) * Ll + lr) * Dd;
#pragma unroll
      for (int j = 0; j < Dd; ++j) dst[j] = (_Float16)(vals[j] * sc);
    } else {
      _Float16* dst = out + (size_t)(bt * Hh + hh) * Dd * Ll + lr;
#pragma unroll
      for (int j = 0; j < Dd; ++j) dst[(size_t)j * Ll] = (_Float16)(vals[j] * sc);
    }
  }
}

// ------------------------------------------------- kernel 2: attention
// Scores are bounded (|s| <= 1/sqrt(32), unit q,k with 1/sqrt(d) folded into
// q), so softmax needs NO max subtraction: exp(s) in [0.84, 1.19]. Row sums
// accumulate per-lane and reduce once at the end.
// grid.x = Ll/128 (128 queries/block), grid.y = bt*h. 4 waves, each owns
// TWO 16-query tiles so every K/V fragment feeds 2 WMMAs.
__global__ __launch_bounds__(128) void attn_kernel(
    const _Float16* __restrict__ qn, const _Float16* __restrict__ kn,
    const _Float16* __restrict__ vt, _Float16* __restrict__ xo) {
  __shared__ __align__(32) _Float16 pbuf[4][2][16][32];  // per-wave P tiles

  const int wave = threadIdx.x >> 5;
  const int lane = threadIdx.x & 31;
  const int hl   = lane >> 4;
  const int l16  = lane & 15;
  const int bth  = blockIdx.y;            // 0..127
  const int bt   = bth >> 3;
  const int h    = bth & 7;
  const int qbase = blockIdx.x * 128 + wave * 32;

  const _Float16* qh = qn + (size_t)bth * Ll * Dd;
  const _Float16* kh = kn + (size_t)bth * Ll * Dd;
  const _Float16* vh = vt + (size_t)bth * Dd * Ll;

  // Two Q A-fragments: invariant across the key loop (d == 32 == WMMA K).
  const v16h aq0 = load_a_f16(qh + (size_t)(qbase + l16) * Dd, 0, hl);
  const v16h aq1 = load_a_f16(qh + (size_t)(qbase + 16 + l16) * Dd, 0, hl);

  v8f o00 = {}, o01 = {}, o10 = {}, o11 = {};
  float lr0[8], lr1[8];
#pragma unroll
  for (int v = 0; v < 8; ++v) { lr0[v] = 0.f; lr1[v] = 0.f; }

  for (int kc = 0; kc < Ll; kc += 32) {
    // K B-fragments: lane = key l16, d-range hl*16..+15 contiguous.
    const v16h bk0 = *(const v16h*)(kh + (size_t)(kc + l16) * Dd + hl * 16);
    const v16h bk1 = *(const v16h*)(kh + (size_t)(kc + 16 + l16) * Dd + hl * 16);
    const v8f z = {};
    const v8f s00 = wmma16(aq0, bk0, z);
    const v8f s01 = wmma16(aq0, bk1, z);
    const v8f s10 = wmma16(aq1, bk0, z);
    const v8f s11 = wmma16(aq1, bk1, z);

    // exp + per-lane row-sum accumulation; P tiles staged for the LDS
    // layout transpose (C layout -> A layout). No reductions here.
#pragma unroll
    for (int v = 0; v < 8; ++v) {
      const float e00 = __expf(s00[v]);
      const float e01 = __expf(s01[v]);
      const float e10 = __expf(s10[v]);
      const float e11 = __expf(s11[v]);
      lr0[v] += e00 + e01;
      lr1[v] += e10 + e11;
      const int r = v + 8 * hl;
      pbuf[wave][0][r][l16]      = (_Float16)e00;
      pbuf[wave][0][r][16 + l16] = (_Float16)e01;
      pbuf[wave][1][r][l16]      = (_Float16)e10;
      pbuf[wave][1][r][16 + l16] = (_Float16)e11;
    }

    // Re-read P as A-fragments (wave-private; compiler inserts dscnt waits).
    v16h ap0, ap1;
    {
      const v8h lo0 = *(const v8h*)&pbuf[wave][0][l16][hl * 16];
      const v8h hi0 = *(const v8h*)&pbuf[wave][0][l16][hl * 16 + 8];
      const v8h lo1 = *(const v8h*)&pbuf[wave][1][l16][hl * 16];
      const v8h hi1 = *(const v8h*)&pbuf[wave][1][l16][hl * 16 + 8];
#pragma unroll
      for (int j = 0; j < 8; ++j) {
        ap0[j] = lo0[j]; ap0[8 + j] = hi0[j];
        ap1[j] = lo1[j]; ap1[8 + j] = hi1[j];
      }
    }

    // V^T B-fragments: lane = d-column, 16 contiguous keys.
    const v16h bv0 = *(const v16h*)(vh + (size_t)l16 * Ll + kc + hl * 16);
    const v16h bv1 = *(const v16h*)(vh + (size_t)(16 + l16) * Ll + kc + hl * 16);
    o00 = wmma16(ap0, bv0, o00);
    o01 = wmma16(ap0, bv1, o01);
    o10 = wmma16(ap1, bv0, o10);
    o11 = wmma16(ap1, bv1, o11);
  }

  // Epilogue: one shuffle-reduction of the row sums, divide, store f16
  // into the merged-head [bt,l,c] layout.
#pragma unroll
  for (int v = 0; v < 8; ++v) {
    float s0 = lr0[v], s1 = lr1[v];
#pragma unroll
    for (int off = 1; off < 16; off <<= 1) {
      s0 += __shfl_xor(s0, off, 16);
      s1 += __shfl_xor(s1, off, 16);
    }
    const float i0 = 1.0f / s0;
    const float i1 = 1.0f / s1;
    const int qg = qbase + v + 8 * hl;
    const size_t base0 = ((size_t)bt * Ll + qg) * Cc + h * Dd;
    const size_t base1 = ((size_t)bt * Ll + qg + 16) * Cc + h * Dd;
    xo[base0 + l16]      = (_Float16)(o00[v] * i0);
    xo[base0 + 16 + l16] = (_Float16)(o01[v] * i0);
    xo[base1 + l16]      = (_Float16)(o10[v] * i1);
    xo[base1 + 16 + l16] = (_Float16)(o11[v] * i1);
  }
}

// ------------------------------------------------- kernel 3: output GEMM
// out = x @ Wm^T + bm + shortcut(q), fp32 out. 64x64 tiles like kernel 1.
__global__ __launch_bounds__(256) void out_proj_kernel(
    const _Float16* __restrict__ xo, const _Float16* __restrict__ Wmh,
    const float* __restrict__ bm, const float* __restrict__ shortcut,
    float* __restrict__ out) {
  const int wave = threadIdx.x >> 5;
  const int lane = threadIdx.x & 31;
  const int hl   = lane >> 4;
  const int l16  = lane & 15;
  const int mt   = wave >> 1;
  const int nh   = wave & 1;
  const int mblk = blockIdx.x;
  const int nblk = blockIdx.y;

  const _Float16* arow = xo + (size_t)(mblk * 64 + mt * 16 + l16) * Cc;
  const int n0 = nblk * 64 + nh * 32 + l16;

  v8f acc0 = {}, acc1 = {};
#pragma unroll
  for (int k0 = 0; k0 < Cc; k0 += 32) {
    const v16h a  = load_a_f16(arow, k0, hl);
    const v16h b0 = *(const v16h*)(Wmh + (size_t)n0 * Cc + k0 + hl * 16);
    const v16h b1 = *(const v16h*)(Wmh + (size_t)(n0 + 16) * Cc + k0 + hl * 16);
    acc0 = wmma16(a, b0, acc0);
    acc1 = wmma16(a, b1, acc1);
  }

#pragma unroll
  for (int nt = 0; nt < 2; ++nt) {
    const int ncol = n0 + nt * 16;
    const float bv = bm[ncol];
    const v8f acc = nt ? acc1 : acc0;
#pragma unroll
    for (int v = 0; v < 8; ++v) {
      const size_t idx = (size_t)(mblk * 64 + mt * 16 + v + 8 * hl) * Cc + ncol;
      out[idx] = acc[v] + bv + shortcut[idx];
    }
  }
}

// ---------------------------------------------------------------- launcher
extern "C" void kernel_launch(void* const* d_in, const int* in_sizes, int n_in,
                              void* d_out, int out_size, void* d_ws, size_t ws_size,
                              hipStream_t stream) {
  (void)in_sizes; (void)n_in; (void)out_size; (void)ws_size;
  const float* q  = (const float*)d_in[0];
  const float* k  = (const float*)d_in[1];
  const float* v  = (const float*)d_in[2];
  const float* Wq = (const float*)d_in[3];
  const float* bq = (const float*)d_in[4];
  const float* Wk = (const float*)d_in[5];
  const float* bk = (const float*)d_in[6];
  const float* Wv = (const float*)d_in[7];
  const float* bv = (const float*)d_in[8];
  const float* Wm = (const float*)d_in[9];
  const float* bm = (const float*)d_in[10];

  // Workspace: 4 f16 activation segments of bt*h*l*d = 4M halves (8 MB) each,
  // then 4 f16 weight matrices of Cc*Cc halves (128 KB) each.
  const size_t SEG = (size_t)BT * Hh * Ll * Dd;
  const size_t WSEG = (size_t)Cc * Cc;
  _Float16* ws  = (_Float16*)d_ws;
  _Float16* qn  = ws;
  _Float16* kn  = ws + SEG;
  _Float16* vt  = ws + 2 * SEG;
  _Float16* xo  = ws + 3 * SEG;
  _Float16* wq16 = ws + 4 * SEG;
  _Float16* wk16 = wq16 + WSEG;
  _Float16* wv16 = wk16 + WSEG;
  _Float16* wm16 = wv16 + WSEG;

  cvt_w_kernel<<<(Cc * Cc) / 256, 256, 0, stream>>>(Wq, Wk, Wv, Wm,
                                                    wq16, wk16, wv16, wm16);

  const dim3 gProj(Mtot / 64, Cc / 64);    // 256 x 4 blocks
  proj_norm_kernel<<<gProj, 256, 0, stream>>>(q, wq16, bq, qn, 0);
  proj_norm_kernel<<<gProj, 256, 0, stream>>>(k, wk16, bk, kn, 1);
  proj_norm_kernel<<<gProj, 256, 0, stream>>>(v, wv16, bv, vt, 2);

  attn_kernel<<<dim3(Ll / 128, BT * Hh), 128, 0, stream>>>(qn, kn, vt, xo);

  out_proj_kernel<<<gProj, 256, 0, stream>>>(xo, wm16, bm, q, (float*)d_out);
}
